// Hybrid_QCNN_65266323030224
// MI455X (gfx1250) — compile-verified, matrix operand order checked by reading, not run on hardware
//
#include <hip/hip_runtime.h>
#include <hip/hip_bf16.h>
#include <math.h>

typedef __attribute__((ext_vector_type(16))) _Float16 v16h;
typedef __attribute__((ext_vector_type(8)))  float    v8f;
typedef __attribute__((ext_vector_type(4)))  int      v4i;

#define PI_F 3.14159265358979323846f

#if defined(__AMDGCN__) && __has_builtin(__builtin_amdgcn_global_load_async_to_lds_b128) && __has_builtin(__builtin_amdgcn_s_wait_asynccnt)
#define HAVE_ASYNC_LDS 1
typedef __attribute__((address_space(1))) v4i* g_v4i_ptr;
typedef __attribute__((address_space(3))) v4i* l_v4i_ptr;
#endif

// ---------------------------------------------------------------------------
// conv1: 3->64ch, 3x3 SAME on 32x32, + ReLU + 2x2 maxpool -> f16 [1024,64,16,16]
// Block (512 thr = 16 waves) = one 8x8 output region x all 64 channels.
// ---------------------------------------------------------------------------
__global__ __launch_bounds__(512) void conv1_relu_pool(
    const float* __restrict__ x, const float* __restrict__ w,
    const float* __restrict__ bias, _Float16* __restrict__ out1)
{
    __shared__ __align__(32) _Float16 simg[378];       // 3*100 patch + zero pad
    __shared__ __align__(32) _Float16 swt[4][32][16];  // frag-packed weights
    __shared__ int aoff[32];                           // [g][h] -> patch offset

    int tid = threadIdx.x;
    int n  = blockIdx.x >> 4;
    int pg = blockIdx.x & 15;                 // 8x8 region within 32x32
    int y0g = (pg >> 2) << 3, x0g = (pg & 3) << 3;

    if (tid < 378) {
        float v = 0.f;
        if (tid < 300) {
            int c = tid / 100, r = tid - c * 100;
            int ry = r / 10, rx = r - ry * 10;
            int iy = y0g - 1 + ry, ix = x0g - 1 + rx;
            if (iy >= 0 && iy < 32 && ix >= 0 && ix < 32)
                v = x[((n * 3 + c) * 32 + iy) * 32 + ix];
        }
        simg[tid] = (_Float16)v;
    }
    for (int i = tid; i < 2048; i += 512) {
        int ct = i >> 9, r = i & 511, l = r >> 4, h = r & 15;
        int K = ((h >> 3) << 4) + ((l >> 4) << 3) + (h & 7);
        float v = 0.f;
        if (K < 27) {
            int c = K / 9, rm = K - c * 9, ky = rm / 3, kx = rm - ky * 3;
            int o = (ct << 4) + (l & 15);
            v = w[((o * 3 + c) * 3 + ky) * 3 + kx];
        }
        swt[ct][l][h] = (_Float16)v;
    }
    if (tid < 32) {
        int g = tid >> 4, h = tid & 15;
        int K = ((h >> 3) << 4) + (g << 3) + (h & 7);
        int off = 300;
        if (K < 27) {
            int c = K / 9, rm = K - c * 9, ky = rm / 3, kx = rm - ky * 3;
            off = c * 100 + ky * 10 + kx;
        }
        aoff[tid] = off;
    }
    __syncthreads();

    int wave = tid >> 5, lane = tid & 31;
    int pt = wave >> 2, ct = wave & 3;        // 4 pos sub-tiles x 4 ch-tiles
    int y0 = y0g + ((pt >> 1) << 2), x0 = x0g + ((pt & 1) << 2);
    int lm = lane & 15, glane = lane >> 4;
    int ly = lm >> 2, lx = lm & 3;
    int yx = ((((pt >> 1) << 2) + ly) * 10) + ((pt & 1) << 2) + lx;

    v16h a, b;
    const int* op = &aoff[glane << 4];
    #pragma unroll
    for (int h = 0; h < 16; ++h) a[h] = simg[op[h] + yx];
    b = *(const v16h*)&swt[ct][lane][0];

    v8f acc = {};
    acc = __builtin_amdgcn_wmma_f32_16x16x32_f16(false, a, false, b,
                                                 (short)0, acc, false, false);
    int ch = (ct << 4) + lm;
    float bs = bias[ch];
    float v[8];
    #pragma unroll
    for (int r = 0; r < 8; ++r) { float t = acc[r] + bs; v[r] = t > 0.f ? t : 0.f; }
    int py = (y0 >> 1) + glane;
    #pragma unroll
    for (int px = 0; px < 2; ++px) {
        float m = fmaxf(fmaxf(v[2*px], v[2*px+1]), fmaxf(v[4+2*px], v[4+2*px+1]));
        out1[((n * 64 + ch) * 16 + py) * 16 + ((x0 >> 1) + px)] = (_Float16)m;
    }
}

// ---------------------------------------------------------------------------
// conv2: 64->16ch, 3x3 SAME on 16x16 (f16 in), + ReLU + pool -> f16 [1024,16,8,8]
// Block (512 thr = 16 waves) = one image; LDS image + frag-packed weights.
// ---------------------------------------------------------------------------
__global__ __launch_bounds__(512) void conv2_relu_pool(
    const _Float16* __restrict__ in, const float* __restrict__ w,
    const float* __restrict__ bias, _Float16* __restrict__ out2)
{
    __shared__ __align__(32) _Float16 simg[64 * 324];   // 64 x 18x18 padded
    __shared__ __align__(32) _Float16 swt[18][32][16];  // frag-packed weights
    __shared__ int aoff[18 * 32];                       // [t][g*16+h] offsets

    int tid = threadIdx.x;
    int n = blockIdx.x;

    for (int i = tid; i < 64 * 324; i += 512) {
        int c = i / 324, r = i - c * 324;
        int ry = r / 18, rx = r - ry * 18;
        int iy = ry - 1, ix = rx - 1;
        float v = 0.f;
        if (iy >= 0 && iy < 16 && ix >= 0 && ix < 16)
            v = (float)in[((n * 64 + c) * 16 + iy) * 16 + ix];
        simg[i] = (_Float16)v;
    }
    for (int i = tid; i < 18 * 512; i += 512) {
        int t = i >> 9, r = i & 511, l = r >> 4, h = r & 15;
        int K = t * 32 + ((h >> 3) << 4) + ((l >> 4) << 3) + (h & 7);
        int c = K / 9, rm = K - c * 9, ky = rm / 3, kx = rm - ky * 3;
        swt[t][l][h] = (_Float16)w[(((l & 15) * 64 + c) * 3 + ky) * 3 + kx];
    }
    for (int i = tid; i < 18 * 32; i += 512) {
        int t = i >> 5, r = i & 31, g = r >> 4, h = r & 15;
        int K = t * 32 + ((h >> 3) << 4) + (g << 3) + (h & 7);
        int c = K / 9, rm = K - c * 9, ky = rm / 3, kx = rm - ky * 3;
        aoff[i] = c * 324 + ky * 18 + kx;
    }
    __syncthreads();

    int wave = tid >> 5, lane = tid & 31;
    int y0 = (wave >> 2) << 2, x0 = (wave & 3) << 2;
    int lm = lane & 15, glane = lane >> 4;
    int ly = lm >> 2, lx = lm & 3;
    int yx = (y0 + ly) * 18 + (x0 + lx);

    v8f acc = {};
    for (int t = 0; t < 18; ++t) {
        v16h a;
        const int* op = &aoff[(t << 5) + (glane << 4)];
        #pragma unroll
        for (int h = 0; h < 16; ++h) a[h] = simg[op[h] + yx];
        v16h b = *(const v16h*)&swt[t][lane][0];
        acc = __builtin_amdgcn_wmma_f32_16x16x32_f16(false, a, false, b,
                                                     (short)0, acc, false, false);
    }
    float bs = bias[lm];
    float v[8];
    #pragma unroll
    for (int r = 0; r < 8; ++r) { float t = acc[r] + bs; v[r] = t > 0.f ? t : 0.f; }
    int py = (y0 >> 1) + glane;
    #pragma unroll
    for (int px = 0; px < 2; ++px) {
        float m = fmaxf(fmaxf(v[2*px], v[2*px+1]), fmaxf(v[4+2*px], v[4+2*px+1]));
        out2[((n * 16 + lm) * 8 + py) * 8 + ((x0 >> 1) + px)] = (_Float16)m;
    }
}

// ---------------------------------------------------------------------------
// conv3 (16->4ch, 8x8) + leaky ReLU + sigmoid*pi -> angles [65536 patches][4]
// Block = 4 images; 8 KB of f16 activations staged to LDS via async copies.
// ---------------------------------------------------------------------------
__global__ __launch_bounds__(256) void conv3_angles(
    const _Float16* __restrict__ in, const float* __restrict__ w,
    const float* __restrict__ bias, float* __restrict__ ang)
{
    __shared__ __align__(16) _Float16 stile[4096];   // 4 imgs x 16ch x 8x8
    int tid = threadIdx.x;
    size_t base = (size_t)blockIdx.x * 4096;
#ifdef HAVE_ASYNC_LDS
    #pragma unroll
    for (int i = 0; i < 2; ++i) {
        int k = tid + (i << 8);                      // 512 chunks of 16B
        __builtin_amdgcn_global_load_async_to_lds_b128(
            (g_v4i_ptr)(unsigned long long)(in + base + (size_t)k * 8),
            (l_v4i_ptr)(unsigned int)(unsigned long long)(stile + k * 8),
            0, 0);
    }
    __builtin_amdgcn_s_wait_asynccnt(0);
    __syncthreads();
#else
    for (int i = tid; i < 4096; i += 256) stile[i] = in[base + i];
    __syncthreads();
#endif
    int p = blockIdx.x * 256 + tid;              // 0..65535
    int ln = (p >> 6) & 3, pos = p & 63, y = pos >> 3, xx = pos & 7;
    const _Float16* img = &stile[ln << 10];
    float acc[4] = {bias[0], bias[1], bias[2], bias[3]};
    for (int c = 0; c < 16; ++c)
        for (int ky = 0; ky < 3; ++ky) {
            int iy = y + ky - 1; if (iy < 0 || iy >= 8) continue;
            for (int kx = 0; kx < 3; ++kx) {
                int ix = xx + kx - 1; if (ix < 0 || ix >= 8) continue;
                float v = (float)img[(c * 8 + iy) * 8 + ix];
                #pragma unroll
                for (int o = 0; o < 4; ++o)
                    acc[o] += v * w[((o * 16 + c) * 3 + ky) * 3 + kx];
            }
        }
    #pragma unroll
    for (int o = 0; o < 4; ++o) {
        float a = acc[o] > 0.f ? acc[o] : 0.01f * acc[o];
        ang[p * 4 + o] = PI_F / (1.f + expf(-a));
    }
}

// ---------------------------------------------------------------------------
// 4-qubit statevector simulation, one patch per lane (16 complex amps in VGPRs)
// ---------------------------------------------------------------------------
__device__ __forceinline__ void apply1(float* sr, float* si, int wq,
    float m00r, float m00i, float m01r, float m01i,
    float m10r, float m10i, float m11r, float m11i)
{
    int stride = 1 << (3 - wq);
    #pragma unroll
    for (int base = 0; base < 16; ++base) {
        if (base & stride) continue;
        int i0 = base, i1 = base | stride;
        float a0r = sr[i0], a0i = si[i0], a1r = sr[i1], a1i = si[i1];
        sr[i0] = m00r*a0r - m00i*a0i + m01r*a1r - m01i*a1i;
        si[i0] = m00r*a0i + m00i*a0r + m01r*a1i + m01i*a1r;
        sr[i1] = m10r*a0r - m10i*a0i + m11r*a1r - m11i*a1i;
        si[i1] = m10r*a0i + m10i*a0r + m11r*a1i + m11i*a1r;
    }
}

__device__ __forceinline__ void cnot_g(float* sr, float* si, int a, int b)
{
    int am = 1 << (3 - a), bm = 1 << (3 - b);
    #pragma unroll
    for (int i = 0; i < 16; ++i) {
        if ((i & am) && !(i & bm)) {
            int j = i | bm;
            float tr = sr[i], ti = si[i];
            sr[i] = sr[j]; si[i] = si[j];
            sr[j] = tr;    si[j] = ti;
        }
    }
}

__device__ __forceinline__ void apply_u3(float* sr, float* si, int wq,
                                         float t, float p, float l)
{
    float sh, ch;  sincosf(0.5f * t, &sh, &ch);
    float sl, cl;  sincosf(l,        &sl, &cl);
    float sp, cp;  sincosf(p,        &sp, &cp);
    float spl, cpl; sincosf(p + l,   &spl, &cpl);
    apply1(sr, si, wq,
           ch,        0.f,
           -cl * sh, -sl * sh,
            cp * sh,  sp * sh,
           cpl * ch, spl * ch);
}

__global__ __launch_bounds__(256) void quantum_kernel(
    const float* __restrict__ ang, const float* __restrict__ qw_u3,
    const float* __restrict__ qw_ang, _Float16* __restrict__ q)
{
    int p = blockIdx.x * 256 + threadIdx.x;
    float sr[16], si[16];
    #pragma unroll
    for (int i = 0; i < 16; ++i) { sr[i] = 0.f; si[i] = 0.f; }
    sr[0] = 1.f;

    #pragma unroll
    for (int wq = 0; wq < 4; ++wq) {           // per-patch RY encoding
        float t = ang[p * 4 + wq];
        float sh, ch; sincosf(0.5f * t, &sh, &ch);
        apply1(sr, si, wq, ch, 0.f, -sh, 0.f, sh, 0.f, ch, 0.f);
    }

    const int pa[4] = {0, 1, 2, 3}, pb[4] = {1, 2, 3, 0};
    #pragma unroll
    for (int layer = 0; layer < 2; ++layer) {
        float u[12], an[3];
        #pragma unroll
        for (int i = 0; i < 12; ++i) u[i] = qw_u3[12 * layer + i];
        #pragma unroll
        for (int i = 0; i < 3; ++i)  an[i] = qw_ang[3 * layer + i];
        #pragma unroll
        for (int pr = 0; pr < 4; ++pr) {
            int a = pa[pr], b = pb[pr];
            apply_u3(sr, si, a, u[0], u[1], u[2]);
            apply_u3(sr, si, b, u[3], u[4], u[5]);
            cnot_g(sr, si, a, b);
            { float s, c; sincosf(0.5f * an[0], &s, &c);
              apply1(sr, si, a, c, 0.f, -s, 0.f, s, 0.f, c, 0.f); }      // RY
            { float s, c; sincosf(0.5f * an[1], &s, &c);
              apply1(sr, si, b, c, -s, 0.f, 0.f, 0.f, 0.f, c, s); }      // RZ
            cnot_g(sr, si, b, a);
            { float s, c; sincosf(0.5f * an[2], &s, &c);
              apply1(sr, si, a, c, 0.f, -s, 0.f, s, 0.f, c, 0.f); }      // RY
            cnot_g(sr, si, a, b);
            apply_u3(sr, si, a, u[6], u[7], u[8]);
            apply_u3(sr, si, b, u[9], u[10], u[11]);
        }
    }
    #pragma unroll
    for (int wq = 0; wq < 4; ++wq) {
        int bit = 3 - wq; float e = 0.f;
        #pragma unroll
        for (int i = 0; i < 16; ++i) {
            float pr2 = sr[i]*sr[i] + si[i]*si[i];
            e += ((i >> bit) & 1) ? -pr2 : pr2;
        }
        q[p * 4 + wq] = (_Float16)e;
    }
}

// ---------------------------------------------------------------------------
// FC layer: Out[M,N] = relu?(A[M,K] @ W[N,K]^T + bias[N]); A is f16,
// Out is f16 for intermediate layers, f32 for the final one.
// ---------------------------------------------------------------------------
__global__ __launch_bounds__(256) void fc_wmma(
    const _Float16* __restrict__ A, const float* __restrict__ W,
    const float* __restrict__ bias, void* __restrict__ Out,
    int N, int K, int ntiles, int total_waves, int do_relu, int out_half)
{
    int wave = (blockIdx.x * 256 + threadIdx.x) >> 5;
    if (wave >= total_waves) return;           // wave-uniform: EXEC stays all-1
    int lane = threadIdx.x & 31;
    int nt = wave % ntiles, mt = wave / ntiles;
    int mbase = mt << 4, nbase = nt << 4;
    int lm  = lane & 15;
    int khi = (lane >> 4) << 3;
    int row = mbase + lm;
    int col = nbase + lm;

    v8f acc = {};
    for (int t = 0; t < K; t += 32) {
        v16h a, b;
        #pragma unroll
        for (int h = 0; h < 16; ++h) {
            int Kg = t + ((h >> 3) << 4) + khi + (h & 7);
            a[h] = A[row * K + Kg];
            b[h] = (_Float16)((col < N) ? W[col * K + Kg] : 0.f);
        }
        acc = __builtin_amdgcn_wmma_f32_16x16x32_f16(false, a, false, b,
                                                     (short)0, acc, false, false);
    }
    if (col < N) {
        float bs = bias[col];
        int g = lane >> 4;
        #pragma unroll
        for (int r = 0; r < 8; ++r) {
            float v = acc[r] + bs;
            if (do_relu) v = v > 0.f ? v : 0.f;
            int idx = (mbase + (g << 3) + r) * N + col;
            if (out_half) ((_Float16*)Out)[idx] = (_Float16)v;
            else          ((float*)Out)[idx] = v;
        }
    }
}

// ---------------------------------------------------------------------------
extern "C" void kernel_launch(void* const* d_in, const int* in_sizes, int n_in,
                              void* d_out, int out_size, void* d_ws, size_t ws_size,
                              hipStream_t stream)
{
    const float* x      = (const float*)d_in[0];
    const float* c1w    = (const float*)d_in[1];
    const float* c1b    = (const float*)d_in[2];
    const float* c2w    = (const float*)d_in[3];
    const float* c2b    = (const float*)d_in[4];
    const float* c3w    = (const float*)d_in[5];
    const float* c3b    = (const float*)d_in[6];
    const float* qw_u3  = (const float*)d_in[7];
    const float* qw_ang = (const float*)d_in[8];
    const float* f1aw   = (const float*)d_in[9];
    const float* f1ab   = (const float*)d_in[10];
    const float* f1bw   = (const float*)d_in[11];
    const float* f1bb   = (const float*)d_in[12];
    const float* f2aw   = (const float*)d_in[13];
    const float* f2ab   = (const float*)d_in[14];
    const float* f2bw   = (const float*)d_in[15];
    const float* f2bb   = (const float*)d_in[16];

    // workspace layout (bytes, all 16B aligned)
    char* ws = (char*)d_ws;
    _Float16* out1 = (_Float16*)ws;                            // 32 MiB
    _Float16* out2 = (_Float16*)(ws + (size_t)33554432);       // 2 MiB
    float*    angles = (float*)(ws + (size_t)35651584);        // 1 MiB
    _Float16* qbuf = (_Float16*)(ws + (size_t)36700160);       // 512 KiB
    _Float16* h1   = (_Float16*)(ws + (size_t)37224448);       // 128 KiB
    _Float16* h1b  = (_Float16*)(ws + (size_t)37355520);       // 256 KiB
    _Float16* h2   = (_Float16*)(ws + (size_t)37617664);       // 128 KiB
    float*    out  = (float*)d_out;                            // 1024x10

    conv1_relu_pool<<<16384, 512, 0, stream>>>(x, c1w, c1b, out1);
    conv2_relu_pool<<<1024, 512, 0, stream>>>(out1, c2w, c2b, out2);
    conv3_angles<<<256, 256, 0, stream>>>(out2, c3w, c3b, angles);
    quantum_kernel<<<256, 256, 0, stream>>>(angles, qw_u3, qw_ang, qbuf);
    fc_wmma<<<32, 256, 0, stream>>>(qbuf, f1aw, f1ab, h1,  64,  256, 4, 256, 1, 1);
    fc_wmma<<<64, 256, 0, stream>>>(h1,   f1bw, f1bb, h1b, 128, 64,  8, 512, 0, 1);
    fc_wmma<<<32, 256, 0, stream>>>(h1b,  f2aw, f2ab, h2,  64,  128, 4, 256, 1, 1);
    fc_wmma<<< 8, 256, 0, stream>>>(h2,   f2bw, f2bb, (void*)out, 10, 64, 1, 64, 0, 0);
}
